// RelationScorer_18468359372873
// MI455X (gfx1250) — matrix operand order repaired and checked by reading, not compiled, not run on hardware
//
#include <hip/hip_runtime.h>
#include <hip/hip_bf16.h>

// ---------------------------------------------------------------------------
// RelationScorer on MI455X (gfx1250, wave32, WMMA).
//
// GEMM core: fused[524288 x 128] @ W1[128 x 64] via v_wmma_f32_16x16x32_bf16.
// ~8.7 GFLOP total vs ~5 MB of unique data -> matrix-core bound; all gathers
// hit L2 (192 MB). Fused rows are built on the fly in LDS (bf16), never
// materializing the 64 MB [E,K,K,D] gather of the reference.
// ---------------------------------------------------------------------------

#define N_TRK 512
#define N_OBJ 1024
#define DIM   32
#define E_CNT 2048
#define M_CNT 8192
#define KCAND 16

typedef __attribute__((ext_vector_type(8)))  float  v8f;
typedef __attribute__((ext_vector_type(16))) __bf16 v16bf;
typedef __attribute__((ext_vector_type(8)))  __bf16 v8bf;

// --------------------------- prep kernels ----------------------------------

// Zero the (lo,hi) -> trk-edge-index table (262144 ints).
__global__ void k_zero_tab(int* __restrict__ tab) {
    tab[blockIdx.x * 256 + threadIdx.x] = 0;
}

// Fill table with (edge index + 1); atomicMax makes duplicate keys deterministic.
__global__ void k_fill_tab(const int* __restrict__ tei, int* __restrict__ tab) {
    int m = blockIdx.x * 256 + threadIdx.x;
    if (m >= M_CNT) return;
    int a = tei[m], b = tei[M_CNT + m];
    int lo = min(a, b), hi = max(a, b);
    atomicMax(&tab[lo * N_TRK + hi], m + 1);
}

// Pre-swizzle W1 (f32 [128][64]) into per-lane bf16 B fragments for
// v_wmma_f32_16x16x32_bf16.  B is 32x16 per WMMA step:
//   lane = (n&15) + 16*(kk>=16),  element t = kk&15   (kk = k within step).
// Fragment f = kb*4 + c stored as 32 lanes x 16 halves (512 halves each).
__global__ void k_prep_b(const float* __restrict__ W1, unsigned short* __restrict__ bfrag) {
    int idx = blockIdx.x * 256 + threadIdx.x;      // 0..8191
    int k = idx >> 6;                              // 0..127
    int n = idx & 63;                              // 0..63
    int kb = k >> 5, kk = k & 31;
    int c  = n >> 4, nr = n & 15;
    int lane = nr + ((kk >= 16) ? 16 : 0);
    int t    = kk & 15;
    __bf16 v = (__bf16)W1[k * 64 + n];
    bfrag[(kb * 4 + c) * 512 + lane * 16 + t] = __builtin_bit_cast(unsigned short, v);
}

// ----------------------------- main kernel ---------------------------------
// One block per object edge e. 8 waves; wave w handles tiles i = w and w+8
// (tile i == first-candidate index; rows j = 0..15 are the second candidate).

#define ROWSTR 136   // LDS row stride in halves (128 + 8 pad -> conflict-free)

__global__ __launch_bounds__(256) void relscore_main(
    const float* __restrict__ objf,   // [E][32]
    const float* __restrict__ trkf,   // [M][32]
    const float* __restrict__ b1,     // [64]
    const float* __restrict__ W2,     // [64]
    const float* __restrict__ b2,     // [1]
    const int*   __restrict__ oidx,   // [2][E]
    const int*   __restrict__ cand,   // [N_OBJ][16]
    const int*   __restrict__ tab,    // [512][512] (lo,hi)->edge+1
    const unsigned short* __restrict__ bfrag, // 8192 bf16 B-fragment halves
    float* __restrict__ out)          // [E][16][16]
{
    __shared__ __attribute__((aligned(16))) __bf16 sB[16 * 512];        // 16 KB
    __shared__ __attribute__((aligned(16))) __bf16 sA[8][16 * ROWSTR];  // 34 KB
    __shared__ int sFlag[8][16];

    const int tid  = threadIdx.x;
    const int lane = tid & 31;
    const int w    = tid >> 5;
    const int e    = blockIdx.x;
    const int ln   = lane & 15;
    const int hs   = lane >> 4;

    // Cooperative copy of B fragments 0..7 to LDS (512 uint4 = 4096 halves).
    {
        const uint4* src = (const uint4*)bfrag;
        uint4*       dst = (uint4*)sB;
        #pragma unroll
        for (int i = 0; i < 2; ++i) dst[tid + 256 * i] = src[tid + 256 * i];
    }

    const int oa = oidx[e];
    const int ob = oidx[E_CNT + e];
    const float o_d = objf[e * DIM + lane];     // this lane's obj feature dim

    float b1v[4], w2v[4];
    #pragma unroll
    for (int c = 0; c < 4; ++c) { b1v[c] = b1[c * 16 + ln]; w2v[c] = W2[c * 16 + ln]; }
    const float bias2 = b2[0];

    __syncthreads();

    for (int tile = w; tile < 16; tile += 8) {
        const int ta = cand[oa * KCAND + tile];
        __bf16* rowbase = sA[w];

        // ---- stage 16 fused rows (bf16) into this wave's LDS region ----
        for (int r = 0; r < 16; ++r) {
            int tb = cand[ob * KCAND + r];
            int lo = min(ta, tb), hi = max(ta, tb);
            int idx = tab[lo * N_TRK + hi];
            float tr = 0.0f;
            if (idx) tr = trkf[(idx - 1) * DIM + lane];
            if (ta > tb && lane < 3) tr = -tr;        // _swap_relation
            float ad = fabsf(o_d - tr);
            float pd = o_d * tr;
            __bf16* rp = rowbase + r * ROWSTR;
            rp[lane]       = (__bf16)o_d;
            rp[32 + lane]  = (__bf16)tr;
            rp[64 + lane]  = (__bf16)ad;
            rp[96 + lane]  = (__bf16)pd;
            if (lane == 0) sFlag[w][r] = (ta == tb) ? 1 : (idx == 0 ? 2 : 0);
        }
        // Same-wave DS ops are in-order: stores above are visible to loads below.

        // ---- 16x64 hidden layer via 16 bf16 WMMAs (K = 128 in 4 steps) ----
        v8f acc[4] = {v8f{}, v8f{}, v8f{}, v8f{}};
        #pragma unroll
        for (int kb = 0; kb < 4; ++kb) {
            // A fragment (16x32 bf16): lane m = ln, halves in contiguous k order (as in k_prep_b):
            // K = kb*32 + hs*16 + t
            const __bf16* ra = rowbase + ln * ROWSTR + kb * 32 + hs * 16;
            v8bf a0 = *(const v8bf*)(ra);
            v8bf a1 = *(const v8bf*)(ra + 8);
            v16bf A = __builtin_shufflevector(a0, a1, 0, 1, 2, 3, 4, 5, 6, 7,
                                                      8, 9, 10, 11, 12, 13, 14, 15);
            #pragma unroll
            for (int c = 0; c < 4; ++c) {
                const __bf16* bb = sB + (kb * 4 + c) * 512 + lane * 16;
                v8bf bb0 = *(const v8bf*)(bb);
                v8bf bb1 = *(const v8bf*)(bb + 8);
                v16bf B = __builtin_shufflevector(bb0, bb1, 0, 1, 2, 3, 4, 5, 6, 7,
                                                            8, 9, 10, 11, 12, 13, 14, 15);
                acc[c] = __builtin_amdgcn_wmma_f32_16x16x32_bf16(
                    false, A, false, B, (short)0, acc[c], false, false);
            }
        }

        // ---- second layer: score = relu(h + b1) @ W2 + b2 ----
        // acc[c] vgpr j at lane L holds h[row = j + 8*hs, col = c*16 + ln].
        float partial[8];
        #pragma unroll
        for (int j = 0; j < 8; ++j) partial[j] = 0.0f;
        #pragma unroll
        for (int c = 0; c < 4; ++c) {
            #pragma unroll
            for (int j = 0; j < 8; ++j) {
                float h = acc[c][j] + b1v[c];
                h = fmaxf(h, 0.0f);
                partial[j] = fmaf(h, w2v[c], partial[j]);
            }
        }
        #pragma unroll
        for (int j = 0; j < 8; ++j) {
            float s = partial[j];
            s += __shfl_xor(s, 1);   // reductions stay inside each 16-lane half
            s += __shfl_xor(s, 2);
            s += __shfl_xor(s, 4);
            s += __shfl_xor(s, 8);
            if (ln == 0) {
                int r = j + 8 * hs;
                int flag = sFlag[w][r];
                float v = (flag == 1) ? -10000.0f
                        : (flag == 2) ? -0.5f
                                      : (s + bias2);
                out[(e * 16 + tile) * 16 + r] = v;
            }
        }
    }
}

// ------------------------------ launcher -----------------------------------

extern "C" void kernel_launch(void* const* d_in, const int* in_sizes, int n_in,
                              void* d_out, int out_size, void* d_ws, size_t ws_size,
                              hipStream_t stream) {
    const float* objf = (const float*)d_in[0];   // obj_edge_feat [E,32]
    const float* trkf = (const float*)d_in[1];   // trk_edge_feat [M,32]
    const float* W1   = (const float*)d_in[2];   // [128,64]
    const float* b1   = (const float*)d_in[3];   // [64]
    const float* W2   = (const float*)d_in[4];   // [64,1]
    const float* b2   = (const float*)d_in[5];   // [1]
    const int*   oix  = (const int*)d_in[6];     // obj_edge_index [2,E]
    const int*   tix  = (const int*)d_in[7];     // trk_edge_index [2,M]
    const int*   cand = (const int*)d_in[8];     // [N_OBJ,16]

    int* tab = (int*)d_ws;                                          // 1 MB
    unsigned short* bfrag = (unsigned short*)((char*)d_ws + (1 << 20)); // 16 KB

    k_zero_tab<<<(N_TRK * N_TRK) / 256, 256, 0, stream>>>(tab);
    k_fill_tab<<<M_CNT / 256, 256, 0, stream>>>(tix, tab);
    k_prep_b<<<(128 * 64) / 256, 256, 0, stream>>>(W1, bfrag);
    relscore_main<<<E_CNT, 256, 0, stream>>>(objf, trkf, b1, W2, b2,
                                             oix, cand, tab, bfrag, (float*)d_out);
}
